// CharNN_16707422781941
// MI455X (gfx1250) — matrix operand order repaired and checked by reading, not compiled, not run on hardware
//
#include <hip/hip_runtime.h>

// ---------------------------------------------------------------------------
// CharNN (2-layer GRU LM) for MI455X / gfx1250 — bf16 WMMA pipeline, v2.1.
//  - Weights pre-packed into WMMA tile-major bf16 (coalesced B-frag loads)
//  - Register double-buffered (software pipelined) K loops
//  - GRU step streams W_hh tiles L2 -> LDS via Tensor Data Mover, double
//    buffered with s_wait_tensorcnt + workgroup barriers
// ---------------------------------------------------------------------------

typedef unsigned short u16;
typedef __attribute__((ext_vector_type(8)))  u16    u16x8;
typedef __attribute__((ext_vector_type(16))) u16    u16x16;
typedef __attribute__((ext_vector_type(16))) __bf16 bf16x16;
typedef __attribute__((ext_vector_type(8)))  float  f32x8;
typedef __attribute__((ext_vector_type(4)))  unsigned int uint32x4;
typedef __attribute__((ext_vector_type(8)))  int    int32x8;
typedef __attribute__((ext_vector_type(4)))  int    int32x4;

#define BB 64
#define TT 512
#define VV 256
#define EE 1024
#define HH 1024
#define LLAYERS 2
#define G3 (3 * HH)

// float -> bf16 round-to-nearest-even
__device__ __forceinline__ u16 f2bf(float f) {
  unsigned u = __builtin_bit_cast(unsigned, f);
  u += 0x7FFFu + ((u >> 16) & 1u);
  return (u16)(u >> 16);
}

// A-matrix fragment (16x32 bf16, MxK). Lane L holds M=L&15, K-runs
// [kb,kb+8) and [kb+16,kb+24), kb=(L<16)?0:8. p = row base + k0 + kb.
__device__ __forceinline__ bf16x16 load_a_frag(const u16* p) {
  u16x16 t;
  ((u16x8*)&t)[0] = ((const u16x8*)p)[0];
  ((u16x8*)&t)[1] = ((const u16x8*)(p + 16))[0];
  return __builtin_bit_cast(bf16x16, t);
}

// B-matrix fragment from a packed 32x16 tile: lane L reads 32 contiguous
// bytes at tile + L*32B (whole wave reads one contiguous 1KB tile).
__device__ __forceinline__ bf16x16 load_b_frag(const u16* p) {
  u16x16 t;
  ((u16x8*)&t)[0] = ((const u16x8*)p)[0];
  ((u16x8*)&t)[1] = ((const u16x8*)p)[1];
  return __builtin_bit_cast(bf16x16, t);
}

__device__ __forceinline__ f32x8 wmma_bf16(bf16x16 a, bf16x16 b, f32x8 c) {
  return __builtin_amdgcn_wmma_f32_16x16x32_bf16(
      false, a, false, b, (short)0, c, false, false);
}

// ---------------------------------------------------------------------------
// Tensor Data Mover: 2D tile load (global -> LDS), data_size = 8B units.
// D# built per CDNA5 ISA §8.3/8.4. Groups 2/3 zero (<=2D tensor).
// Toolchain uses the 6-arg builtin: (g0, g1, g2, g3, g3ext, cpol).
// ---------------------------------------------------------------------------
__device__ __forceinline__ void tdm_load_2d(unsigned lds_off_bytes,
                                            const void* gptr,
                                            unsigned tile_d0_qw,  // dim0 in 8B units
                                            unsigned tile_d1,     // rows
                                            unsigned long d0_stride_qw) {
  unsigned long ga = (unsigned long)gptr;
  uint32x4 g0;
  g0.x = 1u;                                        // count=1, user mode
  g0.y = lds_off_bytes;                             // lds_addr
  g0.z = (unsigned)(ga & 0xFFFFFFFFu);              // global_addr[31:0]
  g0.w = (unsigned)((ga >> 32) & 0x01FFFFFFu) | (2u << 30);  // addr[56:32], type=2

  unsigned w0 = (3u << 16);                         // wg_mask=0, data_size=3 (8B)
  unsigned w1 = (tile_d0_qw & 0xFFFFu) << 16;       // tensor_dim0[15:0]
  unsigned w2 = ((tile_d0_qw >> 16) & 0xFFFFu) | ((tile_d1 & 0xFFFFu) << 16);
  unsigned w3 = ((tile_d1 >> 16) & 0xFFFFu) | ((tile_d0_qw & 0xFFFFu) << 16); // tile_dim0
  unsigned w4 = (tile_d1 & 0xFFFFu);                // tile_dim1; tile_dim2=0
  unsigned w5 = (unsigned)(d0_stride_qw & 0xFFFFFFFFu);      // dim0_stride[31:0]
  unsigned w6 = (unsigned)((d0_stride_qw >> 32) & 0xFFFFu);  // dim0_stride[47:32]
  unsigned w7 = 0u;                                 // dim1_stride unused (tile_dim2=0)
  int32x8 g1;
  g1[0] = (int)w0; g1[1] = (int)w1; g1[2] = (int)w2; g1[3] = (int)w3;
  g1[4] = (int)w4; g1[5] = (int)w5; g1[6] = (int)w6; g1[7] = (int)w7;

  int32x4 z4 = {0, 0, 0, 0};
  int32x8 z8 = {0, 0, 0, 0, 0, 0, 0, 0};
  __builtin_amdgcn_tensor_load_to_lds(g0, g1, z4, z4, z8, 0);
}

// ---------------------------------------------------------------------------
// Prep kernels
// ---------------------------------------------------------------------------

__global__ void k_convert(const float* __restrict__ src, u16* __restrict__ dst, long n) {
  long i = (long)blockIdx.x * blockDim.x + threadIdx.x;
  if (i < n) dst[i] = f2bf(src[i]);
}

// Pack W [N,K] (f32, row-major) into WMMA tile-major bf16:
// Bp[((n0/16)*(K/32) + k0/32)*512 + ln*16 + s] = bf16(W[(n0+ln%16)*K + k0+kk]), kk = (s&7)+8*(ln/16)+16*(s/8)
__global__ void k_pack_b(const float* __restrict__ W, u16* __restrict__ Bp,
                         int N, int K) {
  long i = (long)blockIdx.x * blockDim.x + threadIdx.x;  // output index (coalesced)
  if (i >= (long)N * K) return;
  int tile = (int)(i >> 9);
  int rem = (int)(i & 511);
  int ln = rem >> 4, s = rem & 15;
  int nn = ln & 15, kk = (s & 7) + 8 * (ln >> 4) + 16 * (s >> 3);
  int ktiles = K >> 5;
  int nblk = tile / ktiles, kt = tile % ktiles;
  Bp[i] = f2bf(W[(size_t)(nblk * 16 + nn) * K + (kt * 32 + kk)]);
}

// x_bf16[(b*T+t), e] = emb_bf16[X[b*T+t], e]   (E == 1024)
__global__ void k_embed(const int* __restrict__ X, const u16* __restrict__ embb,
                        u16* __restrict__ x, long n) {
  long i = (long)blockIdx.x * blockDim.x + threadIdx.x;
  if (i >= n) return;
  int tok = X[i >> 10];
  x[i] = embb[((size_t)tok << 10) + (i & 1023)];
}

__global__ void k_hinit(const float* __restrict__ h0l, float* __restrict__ hf,
                        u16* __restrict__ hb, int n) {
  int i = blockIdx.x * blockDim.x + threadIdx.x;
  if (i >= n) return;
  float v = h0l[i];
  hf[i] = v;
  hb[i] = f2bf(v);
}

// ---------------------------------------------------------------------------
// GEMM: C[M,N] = A[M,K](bf16) x Bp (tile-packed bf16) + bias[N], f32 out.
// Block 256 threads (8 waves); block tile 128(M) x 64(N); wave = 16x64.
// K loop software-pipelined (register double buffer).
// ---------------------------------------------------------------------------

__global__ void k_gemm_bias(const u16* __restrict__ A, const u16* __restrict__ Bp,
                            const float* __restrict__ bias, float* __restrict__ C,
                            int N, int K) {
  const int lane = threadIdx.x & 31;
  const int wave = threadIdx.x >> 5;
  const int m0 = blockIdx.x * 128 + wave * 16;
  const int n0 = blockIdx.y * 64;
  const int am = lane & 15;
  const int kb = (lane < 16) ? 0 : 8;
  const int ktiles = K >> 5;
  const u16* Arow = A + (size_t)(m0 + am) * K + kb;
  const u16* Bb[4];
#pragma unroll
  for (int j = 0; j < 4; ++j)
    Bb[j] = Bp + ((size_t)((n0 >> 4) + j) * ktiles) * 512 + lane * 16;

  f32x8 acc[4] = {};
  bf16x16 a_cur = load_a_frag(Arow);
  bf16x16 b_cur[4];
#pragma unroll
  for (int j = 0; j < 4; ++j) b_cur[j] = load_b_frag(Bb[j]);

  for (int kt = 0; kt < ktiles - 1; ++kt) {
    bf16x16 a_nxt = load_a_frag(Arow + (kt + 1) * 32);
    bf16x16 b_nxt[4];
#pragma unroll
    for (int j = 0; j < 4; ++j) b_nxt[j] = load_b_frag(Bb[j] + (size_t)(kt + 1) * 512);
#pragma unroll
    for (int j = 0; j < 4; ++j) acc[j] = wmma_bf16(a_cur, b_cur[j], acc[j]);
    a_cur = a_nxt;
#pragma unroll
    for (int j = 0; j < 4; ++j) b_cur[j] = b_nxt[j];
  }
#pragma unroll
  for (int j = 0; j < 4; ++j) acc[j] = wmma_bf16(a_cur, b_cur[j], acc[j]);

  const int cn = lane & 15;
  const int mo = (lane < 16) ? 0 : 8;
#pragma unroll
  for (int j = 0; j < 4; ++j) {
    int col = n0 + j * 16 + cn;
    float bv = bias[col];
#pragma unroll
    for (int v = 0; v < 8; ++v) {
      int row = m0 + mo + v;
      C[(size_t)row * N + col] = acc[j][v] + bv;
    }
  }
}

// Same GEMM, N=VV, logits[:, :-1] row remap into d_out.
__global__ void k_gemm_logits(const u16* __restrict__ A, const u16* __restrict__ Bp,
                              const float* __restrict__ bias, float* __restrict__ out,
                              int K) {
  const int lane = threadIdx.x & 31;
  const int wave = threadIdx.x >> 5;
  const int m0 = blockIdx.x * 128 + wave * 16;
  const int n0 = blockIdx.y * 64;
  const int am = lane & 15;
  const int kb = (lane < 16) ? 0 : 8;
  const int ktiles = K >> 5;
  const u16* Arow = A + (size_t)(m0 + am) * K + kb;
  const u16* Bb[4];
#pragma unroll
  for (int j = 0; j < 4; ++j)
    Bb[j] = Bp + ((size_t)((n0 >> 4) + j) * ktiles) * 512 + lane * 16;

  f32x8 acc[4] = {};
  bf16x16 a_cur = load_a_frag(Arow);
  bf16x16 b_cur[4];
#pragma unroll
  for (int j = 0; j < 4; ++j) b_cur[j] = load_b_frag(Bb[j]);

  for (int kt = 0; kt < ktiles - 1; ++kt) {
    bf16x16 a_nxt = load_a_frag(Arow + (kt + 1) * 32);
    bf16x16 b_nxt[4];
#pragma unroll
    for (int j = 0; j < 4; ++j) b_nxt[j] = load_b_frag(Bb[j] + (size_t)(kt + 1) * 512);
#pragma unroll
    for (int j = 0; j < 4; ++j) acc[j] = wmma_bf16(a_cur, b_cur[j], acc[j]);
    a_cur = a_nxt;
#pragma unroll
    for (int j = 0; j < 4; ++j) b_cur[j] = b_nxt[j];
  }
#pragma unroll
  for (int j = 0; j < 4; ++j) acc[j] = wmma_bf16(a_cur, b_cur[j], acc[j]);

  const int cn = lane & 15;
  const int mo = (lane < 16) ? 0 : 8;
#pragma unroll
  for (int j = 0; j < 4; ++j) {
    int col = n0 + j * 16 + cn;
    float bv = bias[col];
#pragma unroll
    for (int v = 0; v < 8; ++v) {
      int row = m0 + mo + v;  // = b*T + t
      int b = row >> 9;       // T = 512
      int t = row & (TT - 1);
      if (t < TT - 1)
        out[((size_t)b * (TT - 1) + t) * VV + col] = acc[j][v] + bv;
    }
  }
}

// ---------------------------------------------------------------------------
// GRU time step. Grid = H/16 column blocks; block = 128 threads = 4 waves.
// W_hh tiles for the 3 gates stream L2 -> LDS via TDM, double buffered:
//   chunk = 4 k-tiles (K=128) x 3 gates = 12KB, 2 buffers = 24KB LDS.
// wave0 issues TDM + s_wait_tensorcnt; barriers hand buffers to all waves.
// ---------------------------------------------------------------------------

#define KCH_TILES 4                       // k-tiles per chunk
#define CHUNK_U16 (KCH_TILES * 512)       // 2048 u16 = 4KB per gate per chunk

__global__ void k_gru_step(const u16* __restrict__ h_in_b, const float* __restrict__ h_in_f,
                           const float* __restrict__ gx,   // [B*T, 3H] (includes b_ih)
                           const u16* __restrict__ BpHH,   // packed [3H/16][K/32][512]
                           const float* __restrict__ b_hh, // [3H]
                           u16* __restrict__ h_out_b, float* __restrict__ h_out_f,
                           u16* __restrict__ y,            // [B*T, H] bf16
                           int t) {
  __shared__ u16 smem[2][3 * CHUNK_U16];  // [buf][gate*2048 + ktl*512 + lane*16]

  const int lane = threadIdx.x & 31;
  const int wave = threadIdx.x >> 5;
  const int j0 = blockIdx.x * 16;         // column block within H
  const int jblk = j0 >> 4;
  const int m0 = wave * 16;               // batch block
  const int am = lane & 15;
  const int kb = (lane < 16) ? 0 : 8;
  const int ktiles = HH >> 5;             // 32
  const int nchunks = ktiles / KCH_TILES; // 8

  // Packed source: gate g tile (g*64 + jblk, kt) at ((g*64+jblk)*ktiles + kt)*512.
  const u16* src0 = BpHH + (size_t)jblk * ktiles * 512;          // gate r base
  const unsigned long gate_stride_qw = ((size_t)(HH >> 4) * ktiles * 512 * 2) >> 3;

  // Prologue: TDM chunk 0 -> smem[0]
  if (threadIdx.x < 32) {
    tdm_load_2d((unsigned)(unsigned long)(&smem[0][0]), src0,
                (CHUNK_U16 * 2) >> 3, 3, gate_stride_qw);
  }

  const u16* Arow = h_in_b + (size_t)(m0 + am) * HH + kb;
  bf16x16 a_cur = load_a_frag(Arow);

  f32x8 cr = {}, cz = {}, cn = {};
  for (int c = 0; c < nchunks; ++c) {
    if (threadIdx.x < 32) __builtin_amdgcn_s_wait_tensorcnt(0);
    __syncthreads();  // smem[c&1] ready for all waves
    if (c + 1 < nchunks && threadIdx.x < 32) {
      tdm_load_2d((unsigned)(unsigned long)(&smem[(c + 1) & 1][0]),
                  src0 + (size_t)(c + 1) * CHUNK_U16,
                  (CHUNK_U16 * 2) >> 3, 3, gate_stride_qw);
    }
    const u16* buf = &smem[c & 1][0];
#pragma unroll
    for (int ktl = 0; ktl < KCH_TILES; ++ktl) {
      int ktg = c * KCH_TILES + ktl;
      bf16x16 a_nxt = (ktg + 1 < ktiles) ? load_a_frag(Arow + (ktg + 1) * 32) : a_cur;
      const u16* p = buf + ktl * 512 + lane * 16;
      cr = wmma_bf16(a_cur, load_b_frag(p), cr);
      cz = wmma_bf16(a_cur, load_b_frag(p + CHUNK_U16), cz);
      cn = wmma_bf16(a_cur, load_b_frag(p + 2 * CHUNK_U16), cn);
      a_cur = a_nxt;
    }
    __syncthreads();  // all waves done reading smem[c&1] before TDM reuses it
  }

  const int nn = lane & 15;
  const int mo = (lane < 16) ? 0 : 8;
  const int col = j0 + nn;
  const float bhr = b_hh[col];
  const float bhz = b_hh[HH + col];
  const float bhn = b_hh[2 * HH + col];

#pragma unroll
  for (int v = 0; v < 8; ++v) {
    int b = m0 + mo + v;
    const float* gxr = gx + ((size_t)b * TT + t) * G3;
    float r = 1.f / (1.f + __expf(-(gxr[col] + cr[v] + bhr)));
    float z = 1.f / (1.f + __expf(-(gxr[HH + col] + cz[v] + bhz)));
    float n = tanhf(gxr[2 * HH + col] + r * (cn[v] + bhn));
    float hprev = h_in_f[(size_t)b * HH + col];
    float hnew = (1.f - z) * n + z * hprev;
    h_out_f[(size_t)b * HH + col] = hnew;
    u16 hb = f2bf(hnew);
    h_out_b[(size_t)b * HH + col] = hb;
    y[((size_t)b * TT + t) * HH + col] = hb;
  }
}

// ---------------------------------------------------------------------------
// Host orchestration
// ---------------------------------------------------------------------------

extern "C" void kernel_launch(void* const* d_in, const int* in_sizes, int n_in,
                              void* d_out, int out_size, void* d_ws, size_t ws_size,
                              hipStream_t stream) {
  (void)in_sizes; (void)n_in; (void)out_size; (void)ws_size;

  const int*   X     = (const int*)d_in[0];
  const float* h0    = (const float*)d_in[1];
  const float* emb   = (const float*)d_in[2];
  const float* W_ih  = (const float*)d_in[3];
  const float* W_hh  = (const float*)d_in[4];
  const float* b_ih  = (const float*)d_in[5];
  const float* b_hh  = (const float*)d_in[6];
  const float* W_out = (const float*)d_in[7];
  const float* b_out = (const float*)d_in[8];
  float* out = (float*)d_out;

  char* ws = (char*)d_ws;
  size_t off = 0;
  auto take = [&](size_t bytes) -> void* {
    void* p = ws + off;
    off += (bytes + 255) & ~(size_t)255;
    return p;
  };

  u16*   embb  = (u16*)take((size_t)VV * EE * 2);
  u16*   WihP  = (u16*)take((size_t)LLAYERS * EE * G3 * 2);
  u16*   WhhP  = (u16*)take((size_t)LLAYERS * EE * G3 * 2);
  u16*   WoutP = (u16*)take((size_t)HH * VV * 2);
  u16*   xb0   = (u16*)take((size_t)BB * TT * EE * 2);
  u16*   xb1   = (u16*)take((size_t)BB * TT * EE * 2);
  float* gx    = (float*)take((size_t)BB * TT * G3 * 4);
  float* hf0   = (float*)take((size_t)BB * HH * 4);
  float* hf1   = (float*)take((size_t)BB * HH * 4);
  u16*   hbb0  = (u16*)take((size_t)BB * HH * 2);
  u16*   hbb1  = (u16*)take((size_t)BB * HH * 2);

  // --- one-time weight conversion into WMMA tile-packed bf16 ---
  k_convert<<<(VV * EE) / 256, 256, 0, stream>>>(emb, embb, (long)VV * EE);
  for (int l = 0; l < LLAYERS; ++l) {
    k_pack_b<<<((long)G3 * HH) / 256, 256, 0, stream>>>(
        W_ih + (size_t)l * G3 * HH, WihP + (size_t)l * EE * G3, G3, HH);
    k_pack_b<<<((long)G3 * HH) / 256, 256, 0, stream>>>(
        W_hh + (size_t)l * G3 * HH, WhhP + (size_t)l * EE * G3, G3, HH);
  }
  k_pack_b<<<((long)VV * HH) / 256, 256, 0, stream>>>(W_out, WoutP, VV, HH);

  // --- embedding gather ---
  k_embed<<<(unsigned)(((long)BB * TT * EE) / 256), 256, 0, stream>>>(
      X, embb, xb0, (long)BB * TT * EE);

  // --- GRU layers ---
  u16* xcur = xb0;
  u16* xnext = xb1;
  float* hf[2] = {hf0, hf1};
  u16*   hb[2] = {hbb0, hbb1};

  for (int l = 0; l < LLAYERS; ++l) {
    // gx = x @ W_ih^T + b_ih : M=32768, N=3072, K=1024
    k_gemm_bias<<<dim3((BB * TT) / 128, G3 / 64), 256, 0, stream>>>(
        xcur, WihP + (size_t)l * EE * G3, b_ih + (size_t)l * G3, gx, G3, EE);

    k_hinit<<<(BB * HH) / 256, 256, 0, stream>>>(h0 + (size_t)l * BB * HH, hf[0], hb[0],
                                                 BB * HH);

    for (int t = 0; t < TT; ++t) {
      int i = t & 1, o = i ^ 1;
      k_gru_step<<<HH / 16, 128, 0, stream>>>(
          hb[i], hf[i], gx, WhhP + (size_t)l * EE * G3, b_hh + (size_t)l * G3,
          hb[o], hf[o], xnext, t);
    }

    u16* tmp = xcur; xcur = xnext; xnext = tmp;
  }

  // --- logits = y2 @ W_out^T + b_out, rows with t < T-1 only ---
  k_gemm_logits<<<dim3((BB * TT) / 128, VV / 64), 256, 0, stream>>>(
      xcur, WoutP, b_out, out, HH);
}
